// AnatomyGAT_35897336660282
// MI455X (gfx1250) — matrix operand order repaired and checked by reading, not compiled, not run on hardware
//
#include <hip/hip_runtime.h>
#include <math.h>

typedef __attribute__((ext_vector_type(16))) _Float16 v16h;
typedef __attribute__((ext_vector_type(8)))  float    v8f;

#define NN    49152
#define GG    1024
#define FEAT  384
#define NHEAD 8
#define HDIM  48

// ---------------------------------------------------------------- fill
__global__ void fill_f32(float* __restrict__ p, float v, long n) {
  long i  = (long)blockIdx.x * blockDim.x + threadIdx.x;
  long st = (long)gridDim.x * blockDim.x;
  for (; i < n; i += st) p[i] = v;
}

// ------------------------------------------- B-fragment pre-pack (f32 -> f16, WMMA layout)
// Layout: [kblk][ntile][lane][16] f16; zero-filled outside K/Nout so the GEMM
// main loop needs no guards at all.
__global__ void pack_b(const float* __restrict__ B, int ldb, int K, int Nout,
                       int Ntiles, int Kblks, _Float16* __restrict__ out) {
  int idx = blockIdx.x * blockDim.x + threadIdx.x;
  int total = Kblks * Ntiles * 32;
  if (idx >= total) return;
  int lane = idx & 31;
  int tmp  = idx >> 5;
  int nt   = tmp % Ntiles;
  int kblk = tmp / Ntiles;
  int hf = lane >> 4, lr = lane & 15;
  int nc = nt * 16 + lr;
  _Float16* dst = out + (size_t)idx * 16;
#pragma unroll
  for (int i = 0; i < 16; ++i) {
    int kb = kblk * 32 + ((i >> 3) << 4) + hf * 8 + (i & 7);   // ISA B layout: k = 16*(i/8) + half*8 + i%8
    float v = (kb < K && nc < Nout) ? B[(size_t)kb * ldb + nc] : 0.f;
    dst[i] = (_Float16)v;
  }
}

// ------------------------------------------- WMMA GEMM (f16 in, f32 acc)
// C[M,Nout] = act(A[M,K] @ B[K,Nout] + bias).
// One wave32 computes a 16x64 strip (4 N-tiles): the A fragment and its
// f32->f16 conversions are amortized over 4 v_wmma issues per K-step.
// Requires Ntiles % 4 == 0 (guaranteed by callers via padding).
__global__ void gemm_f16_wmma(const float* __restrict__ A, int lda,
                              const _Float16* __restrict__ Bp,
                              const float* __restrict__ bias,
                              float* __restrict__ C, int ldc,
                              int Kp, int K, int Ntiles, int tiles,
                              int Nout, int relu)
{
  int wave = blockIdx.x * (blockDim.x >> 5) + (threadIdx.x >> 5);
  if (wave >= tiles) return;              // wave-uniform exit (EXEC all-1 for WMMA)
  int lane = threadIdx.x & 31;
  int hf   = lane >> 4;                   // wave half
  int lr   = lane & 15;
  int ngroups = Ntiles >> 2;
  int mt  = wave / ngroups;
  int ng  = wave - mt * ngroups;
  int m0  = mt * 16;
  int nt0 = ng << 2;

  const float* arow = A + (size_t)(m0 + lr) * lda;
  const v16h*  bptr = (const v16h*)Bp + (size_t)nt0 * 32 + lane;
  const size_t bstep = (size_t)Ntiles * 32;   // v16h units per k-block

  v8f acc[4] = {{}, {}, {}, {}};
  if (((K & 31) == 0) && ((lda & 3) == 0)) {
    // fast path: K multiple of 32, rows 16B-aligned -> b128 loads, no guards
    for (int k0 = 0; k0 < Kp; k0 += 32) {
      const float* p = arow + k0 + hf * 8;
      float4 a0 = *(const float4*)(p);
      float4 a1 = *(const float4*)(p + 4);
      float4 a2 = *(const float4*)(p + 16);
      float4 a3 = *(const float4*)(p + 20);
      v16h af;
      af[0]=(_Float16)a0.x; af[1]=(_Float16)a0.y; af[2]=(_Float16)a0.z; af[3]=(_Float16)a0.w;
      af[4]=(_Float16)a1.x; af[5]=(_Float16)a1.y; af[6]=(_Float16)a1.z; af[7]=(_Float16)a1.w;
      af[8]=(_Float16)a2.x; af[9]=(_Float16)a2.y; af[10]=(_Float16)a2.z; af[11]=(_Float16)a2.w;
      af[12]=(_Float16)a3.x; af[13]=(_Float16)a3.y; af[14]=(_Float16)a3.z; af[15]=(_Float16)a3.w;
#pragma unroll
      for (int j = 0; j < 4; ++j) {
        v16h bf = bptr[(size_t)j * 32];
        acc[j] = __builtin_amdgcn_wmma_f32_16x16x32_f16(false, af, false, bf,
                                                        (short)0, acc[j], false, false);
      }
      bptr += bstep;
    }
  } else {
    // ragged-K path: clamp addresses (always valid); padded-zero B kills
    // contributions from k >= K, so no value masking needed.
    int Km1 = K - 1;
    for (int k0 = 0; k0 < Kp; k0 += 32) {
      v16h af;
#pragma unroll
      for (int i = 0; i < 16; ++i) {
        int ka = k0 + ((i >> 3) << 4) + hf * 8 + (i & 7);   // ISA A layout
        af[i] = (_Float16)arow[ka < Km1 ? ka : Km1];
      }
#pragma unroll
      for (int j = 0; j < 4; ++j) {
        v16h bf = bptr[(size_t)j * 32];
        acc[j] = __builtin_amdgcn_wmma_f32_16x16x32_f16(false, af, false, bf,
                                                        (short)0, acc[j], false, false);
      }
      bptr += bstep;
    }
  }

#pragma unroll
  for (int j = 0; j < 4; ++j) {
    int nc = (nt0 + j) * 16 + lr;
    if (nc < Nout) {
      float bv = bias ? bias[nc] : 0.f;
#pragma unroll
      for (int vi = 0; vi < 8; ++vi) {    // C/D: row = v + 8*half, col = lane%16
        int row = m0 + vi + hf * 8;
        float val = acc[j][vi] + bv;
        if (relu) val = fmaxf(val, 0.f);
        C[(size_t)row * ldc + nc] = val;
      }
    }
  }
}

// ------------------------------------------- global LayerNorm (mode='graph', batch=None)
__global__ void enc_stats(const float* __restrict__ h0, int c0,
                          float* __restrict__ st, long total) {
  __shared__ float s1[256], s2[256];
  float a = 0.f, b = 0.f;
  for (long i = (long)blockIdx.x * blockDim.x + threadIdx.x; i < total;
       i += (long)gridDim.x * blockDim.x) {
    long n = i >> 7;  int c = (int)(i & 127);
    float v = h0[(size_t)n * FEAT + c0 + c];
    a += v;  b += v * v;
  }
  s1[threadIdx.x] = a; s2[threadIdx.x] = b; __syncthreads();
  for (int o = 128; o > 0; o >>= 1) {
    if ((int)threadIdx.x < o) { s1[threadIdx.x] += s1[threadIdx.x + o];
                                s2[threadIdx.x] += s2[threadIdx.x + o]; }
    __syncthreads();
  }
  if (threadIdx.x == 0) { atomicAdd(&st[0], s1[0]); atomicAdd(&st[1], s2[0]); }
}

__global__ void enc_norm(float* __restrict__ h0, int c0,
                         const float* __restrict__ st,
                         const float* __restrict__ lw, const float* __restrict__ lb,
                         long total) {
  long i = (long)blockIdx.x * blockDim.x + threadIdx.x;
  if (i >= total) return;
  float M    = (float)total;
  float mean = st[0] / M;
  float var  = fmaxf(st[1] / M - mean * mean, 0.f);
  float sd   = sqrtf(var);
  long n = i >> 7;  int c = (int)(i & 127);
  float* p = &h0[(size_t)n * FEAT + c0 + c];
  *p = (*p - mean) / (sd + 1e-5f) * lw[c] + lb[c];
}

// ------------------------------------------- qi/kj: xt[r][n] . q|k[:,h]
__global__ void qk_kernel(const float* __restrict__ xt,
                          const float* __restrict__ q, const float* __restrict__ k,
                          float* __restrict__ qi, float* __restrict__ kj) {
  int idx = blockIdx.x * blockDim.x + threadIdx.x;
  const int total = 3 * NN * NHEAD;
  if (idx >= total) return;
  int h  = idx & 7;
  int nr = idx >> 3;
  int n  = nr % NN;
  int r  = nr / NN;
  const float* row = xt + ((size_t)r * NN + n) * FEAT;
  float sq = 0.f, sk = 0.f;
  for (int c = 0; c < FEAT; ++c) {
    float x = row[c];
    sq += x * q[c * NHEAD + h];
    sk += x * k[c * NHEAD + h];
  }
  qi[idx] = sq; kj[idx] = sk;
}

// ------------------------------------------- edge passes
__device__ inline void atomicMaxF(float* addr, float v) {
  if (v >= 0.f) atomicMax((int*)addr, __float_as_int(v));
  else          atomicMin((unsigned int*)addr, __float_as_uint(v));
}

__device__ inline void edge_decode(int e, int E,
                                   const int* so, const int* dso,
                                   const int* sa, const int* dsa,
                                   const int* ss, const int* dss,
                                   int& t, int& sn, int& dn) {
  t = e / E; int el = e - t * E;
  if (t == 0)      { sn = so[el]; dn = dso[el]; }
  else if (t == 1) { sn = sa[el]; dn = dsa[el]; }
  else             { sn = ss[el]; dn = dss[el]; }
}

__global__ void edge_alpha(const int* so, const int* dso, const int* sa, const int* dsa,
                           const int* ss, const int* dss,
                           const float* __restrict__ qi, const float* __restrict__ kj,
                           float* __restrict__ alp, float* __restrict__ mbuf, int E) {
  int idx = blockIdx.x * blockDim.x + threadIdx.x;
  int total = 3 * E * NHEAD;
  if (idx >= total) return;
  int h = idx & 7, e = idx >> 3, t, sn, dn;
  edge_decode(e, E, so, dso, sa, dsa, ss, dss, t, sn, dn);
  float a = qi[((size_t)t * NN + dn) * NHEAD + h] + kj[((size_t)t * NN + sn) * NHEAD + h];
  a = (a > 0.f) ? a : 0.2f * a;           // leaky_relu(0.2)
  alp[idx] = a;
  atomicMaxF(&mbuf[(size_t)dn * NHEAD + h], a);
}

__global__ void edge_exp(const int* so, const int* dso, const int* sa, const int* dsa,
                         const int* ss, const int* dss,
                         float* __restrict__ alp, const float* __restrict__ mbuf,
                         float* __restrict__ sbuf, int E) {
  int idx = blockIdx.x * blockDim.x + threadIdx.x;
  int total = 3 * E * NHEAD;
  if (idx >= total) return;
  int h = idx & 7, e = idx >> 3, t, sn, dn;
  edge_decode(e, E, so, dso, sa, dsa, ss, dss, t, sn, dn);
  float a = expf(alp[idx] - mbuf[(size_t)dn * NHEAD + h]);
  alp[idx] = a;
  atomicAdd(&sbuf[(size_t)dn * NHEAD + h], a);
}

__global__ void edge_scatter(const int* so, const int* dso, const int* sa, const int* dsa,
                             const int* ss, const int* dss,
                             const float* __restrict__ alp, const float* __restrict__ sbuf,
                             const float* __restrict__ xt, float* __restrict__ rgat, int E) {
  int idx = blockIdx.x * blockDim.x + threadIdx.x;
  int total = 3 * E * NHEAD;
  if (idx >= total) return;
  int h = idx & 7, e = idx >> 3, t, sn, dn;
  edge_decode(e, E, so, dso, sa, dsa, ss, dss, t, sn, dn);
  float w = alp[idx] / (sbuf[(size_t)dn * NHEAD + h] + 1e-16f);
  const float* oj = xt + ((size_t)t * NN + sn) * FEAT + h * HDIM;
  float* outp     = rgat + (size_t)dn * FEAT + h * HDIM;
#pragma unroll 4
  for (int c = 0; c < HDIM; ++c) atomicAdd(&outp[c], w * oj[c]);
}

// ------------------------------------------- residual + elu + per-graph stats
__global__ void node_update(const float* __restrict__ hin, float* __restrict__ rgat,
                            const float* __restrict__ bias, const int* __restrict__ batch,
                            float* __restrict__ gsum, float* __restrict__ gsumsq,
                            float* __restrict__ gcnt) {
  int n = blockIdx.x, t = threadIdx.x;     // blockDim = 128
  int g = batch[n];
  float ls = 0.f, ls2 = 0.f;
  for (int c = t; c < FEAT; c += 128) {
    size_t i = (size_t)n * FEAT + c;
    float v = rgat[i] + bias[c];
    v = (v > 0.f) ? v : (expf(v) - 1.f);   // elu
    v += hin[i];
    rgat[i] = v;                           // rgat reused as pre-norm buffer
    ls += v; ls2 += v * v;
  }
  __shared__ float s1[128], s2[128];
  s1[t] = ls; s2[t] = ls2; __syncthreads();
  for (int o = 64; o > 0; o >>= 1) {
    if (t < o) { s1[t] += s1[t + o]; s2[t] += s2[t + o]; }
    __syncthreads();
  }
  if (t == 0) {
    atomicAdd(&gsum[g], s1[0]);
    atomicAdd(&gsumsq[g], s2[0]);
    atomicAdd(&gcnt[g], (float)FEAT);
  }
}

__global__ void graph_stats(const float* __restrict__ gsum, const float* __restrict__ gsumsq,
                            const float* __restrict__ gcnt,
                            float* __restrict__ gmean, float* __restrict__ grstd) {
  int g = blockIdx.x * blockDim.x + threadIdx.x;
  if (g >= GG) return;
  float cnt  = fmaxf(gcnt[g], (float)FEAT);
  float mean = gsum[g] / cnt;
  float var  = fmaxf(gsumsq[g] / cnt - mean * mean, 0.f);
  gmean[g] = mean;
  grstd[g] = 1.f / sqrtf(var + 1e-5f);
}

__global__ void graph_norm(const float* __restrict__ hraw, const int* __restrict__ batch,
                           const float* __restrict__ gmean, const float* __restrict__ grstd,
                           const float* __restrict__ nw, const float* __restrict__ nb,
                           float* __restrict__ hout) {
  long i = (long)blockIdx.x * blockDim.x + threadIdx.x;
  const long total = (long)NN * FEAT;
  if (i >= total) return;
  long n = i / FEAT;  int c = (int)(i - n * FEAT);
  int g = batch[n];
  hout[i] = (hraw[i] - gmean[g]) * grstd[g] * nw[c] + nb[c];
}

// =================================================================== host
static void pack_gemm(const float* A, int lda, const float* B, int ldb,
                      const float* bias, float* C, int ldc,
                      int M, int K, int Nout, int relu,
                      _Float16* bp, hipStream_t st) {
  int Ntiles = (Nout + 15) / 16;
  if (Ntiles & 3) Ntiles = (Ntiles + 3) & ~3;     // pad so Ntiles % 4 == 0
  int Kblks  = (K + 31) / 32;
  int ptotal = Kblks * Ntiles * 32;
  pack_b<<<(ptotal + 255) / 256, 256, 0, st>>>(B, ldb, K, Nout, Ntiles, Kblks, bp);
  int tiles = (Ntiles >> 2) * ((M + 15) / 16);    // one wave per 16x64 strip
  gemm_f16_wmma<<<(tiles + 7) / 8, 256, 0, st>>>(A, lda, bp, bias, C, ldc,
                                                 Kblks * 32, K, Ntiles, tiles, Nout, relu);
}

static void run_layer(const float* hin, const float* rW, const float* q, const float* k,
                      const float* b, const float* nw, const float* nb,
                      const int* so, const int* dso, const int* sa, const int* dsa,
                      const int* ss, const int* dss, const int* batch, int E,
                      float* xt, float* qi, float* kj, float* alp,
                      float* mbuf, float* sbuf, float* rgat,
                      float* gsum, float* gsumsq, float* gcnt, float* gmean, float* grstd,
                      _Float16* bp_r, float* hout, hipStream_t st) {
  const size_t rpack = (size_t)12 * 24 * 32 * 16;   // Kblks*Ntiles*32 lanes*16 f16
  for (int r = 0; r < 3; ++r)
    pack_gemm(hin, FEAT, rW + (size_t)r * FEAT * FEAT, FEAT, nullptr,
              xt + (size_t)r * NN * FEAT, FEAT, NN, FEAT, FEAT, 0,
              bp_r + (size_t)r * rpack, st);

  int qk_total = 3 * NN * NHEAD;
  qk_kernel<<<(qk_total + 255) / 256, 256, 0, st>>>(xt, q, k, qi, kj);

  fill_f32<<<1024, 256, 0, st>>>(mbuf, -INFINITY, (long)NN * NHEAD);
  fill_f32<<<1024, 256, 0, st>>>(sbuf, 0.f, (long)NN * NHEAD);
  fill_f32<<<2048, 256, 0, st>>>(rgat, 0.f, (long)NN * FEAT);
  fill_f32<<<16, 256, 0, st>>>(gsum, 0.f, (long)3 * GG);   // gsum,gsumsq,gcnt contiguous

  int eblocks = (3 * E * NHEAD + 255) / 256;
  edge_alpha  <<<eblocks, 256, 0, st>>>(so, dso, sa, dsa, ss, dss, qi, kj, alp, mbuf, E);
  edge_exp    <<<eblocks, 256, 0, st>>>(so, dso, sa, dsa, ss, dss, alp, mbuf, sbuf, E);
  edge_scatter<<<eblocks, 256, 0, st>>>(so, dso, sa, dsa, ss, dss, alp, sbuf, xt, rgat, E);

  node_update<<<NN, 128, 0, st>>>(hin, rgat, b, batch, gsum, gsumsq, gcnt);
  graph_stats<<<(GG + 255) / 256, 256, 0, st>>>(gsum, gsumsq, gcnt, gmean, grstd);
  graph_norm<<<(int)(((long)NN * FEAT + 255) / 256), 256, 0, st>>>(
      rgat, batch, gmean, grstd, nw, nb, hout);
}

extern "C" void kernel_launch(void* const* d_in, const int* in_sizes, int n_in,
                              void* d_out, int out_size, void* d_ws, size_t ws_size,
                              hipStream_t stream) {
  (void)n_in; (void)out_size; (void)ws_size;
  const float* x_visual = (const float*)d_in[0];
  const float* x_geom   = (const float*)d_in[1];
  const float* x_prior  = (const float*)d_in[2];
  const float* vis_w  = (const float*)d_in[3];
  const float* vis_b  = (const float*)d_in[4];
  const float* vis_lw = (const float*)d_in[5];
  const float* vis_lb = (const float*)d_in[6];
  const float* geom_w  = (const float*)d_in[7];
  const float* geom_b  = (const float*)d_in[8];
  const float* geom_lw = (const float*)d_in[9];
  const float* geom_lb = (const float*)d_in[10];
  const float* prior_w  = (const float*)d_in[11];
  const float* prior_b  = (const float*)d_in[12];
  const float* prior_lw = (const float*)d_in[13];
  const float* prior_lb = (const float*)d_in[14];
  const float* r1_w = (const float*)d_in[15];
  const float* r1_q = (const float*)d_in[16];
  const float* r1_k = (const float*)d_in[17];
  const float* r1_b = (const float*)d_in[18];
  const float* n1_w = (const float*)d_in[19];
  const float* n1_b = (const float*)d_in[20];
  const float* r2_w = (const float*)d_in[21];
  const float* r2_q = (const float*)d_in[22];
  const float* r2_k = (const float*)d_in[23];
  const float* r2_b = (const float*)d_in[24];
  const float* n2_w = (const float*)d_in[25];
  const float* n2_b = (const float*)d_in[26];
  const float* c_w1 = (const float*)d_in[27];
  const float* c_b1 = (const float*)d_in[28];
  const float* c_w2 = (const float*)d_in[29];
  const float* c_b2 = (const float*)d_in[30];
  const int* eo    = (const int*)d_in[31];
  const int* ea    = (const int*)d_in[32];
  const int* es    = (const int*)d_in[33];
  const int* batch = (const int*)d_in[34];
  const int E = in_sizes[31] / 2;

  // --- workspace layout (f32 region, then 32B-aligned f16 pack region) ---
  float* ws = (float*)d_ws;
  size_t off = 0;
  float* h0   = ws + off; off += (size_t)NN * FEAT;
  float* h1   = ws + off; off += (size_t)NN * FEAT;
  float* xt   = ws + off; off += (size_t)3 * NN * FEAT;
  float* qi   = ws + off; off += (size_t)3 * NN * NHEAD;
  float* kj   = ws + off; off += (size_t)3 * NN * NHEAD;
  float* alp  = ws + off; off += (size_t)3 * E * NHEAD;
  float* mbuf = ws + off; off += (size_t)NN * NHEAD;
  float* sbuf = ws + off; off += (size_t)NN * NHEAD;
  float* rgat = ws + off; off += (size_t)NN * FEAT;
  float* estats = ws + off; off += 8;
  float* gsum   = ws + off; off += GG;   // gsum, gsumsq, gcnt contiguous
  float* gsumsq = ws + off; off += GG;
  float* gcnt   = ws + off; off += GG;
  float* gmean  = ws + off; off += GG;
  float* grstd  = ws + off; off += GG;
  off = (off + 7) & ~(size_t)7;          // 32B align for v16h fragment loads
  _Float16* pk = (_Float16*)(ws + off);
  size_t poff = 0;
  _Float16* bp_vis   = pk + poff; poff += (size_t)32 * 8 * 512;  // Kblks*Ntiles*512
  _Float16* bp_geom  = pk + poff; poff += (size_t)1 * 8 * 512;
  _Float16* bp_prior = pk + poff; poff += (size_t)2 * 8 * 512;
  _Float16* bp_r     = pk + poff; poff += (size_t)3 * 12 * 24 * 512;
  _Float16* bp_c1    = pk + poff; poff += (size_t)12 * 8 * 512;
  _Float16* bp_c2    = pk + poff; poff += (size_t)4 * 4 * 512;
  float* h2     = h0;   // h0 dead after layer-1 residual
  float* hidden = xt;   // xt dead after layer-2 edge passes

  // --- encoders: relu(x@W+b), then global LN per 128-col block ---
  fill_f32<<<1, 256, 0, stream>>>(estats, 0.f, 8);
  pack_gemm(x_visual, 1024, vis_w,   128, vis_b,   h0 + 0,   FEAT, NN, 1024, 128, 1, bp_vis,   stream);
  pack_gemm(x_geom,   6,    geom_w,  128, geom_b,  h0 + 128, FEAT, NN, 6,    128, 1, bp_geom,  stream);
  pack_gemm(x_prior,  50,   prior_w, 128, prior_b, h0 + 256, FEAT, NN, 50,   128, 1, bp_prior, stream);
  long etotal = (long)NN * 128;
  enc_stats<<<1024, 256, 0, stream>>>(h0, 0,   estats + 0, etotal);
  enc_stats<<<1024, 256, 0, stream>>>(h0, 128, estats + 2, etotal);
  enc_stats<<<1024, 256, 0, stream>>>(h0, 256, estats + 4, etotal);
  int nblk = (int)((etotal + 255) / 256);
  enc_norm<<<nblk, 256, 0, stream>>>(h0, 0,   estats + 0, vis_lw,   vis_lb,   etotal);
  enc_norm<<<nblk, 256, 0, stream>>>(h0, 128, estats + 2, geom_lw,  geom_lb,  etotal);
  enc_norm<<<nblk, 256, 0, stream>>>(h0, 256, estats + 4, prior_lw, prior_lb, etotal);

  // --- RGAT layers ---
  run_layer(h0, r1_w, r1_q, r1_k, r1_b, n1_w, n1_b,
            eo, eo + E, ea, ea + E, es, es + E, batch, E,
            xt, qi, kj, alp, mbuf, sbuf, rgat,
            gsum, gsumsq, gcnt, gmean, grstd, bp_r, h1, stream);
  run_layer(h1, r2_w, r2_q, r2_k, r2_b, n2_w, n2_b,
            eo, eo + E, ea, ea + E, es, es + E, batch, E,
            xt, qi, kj, alp, mbuf, sbuf, rgat,
            gsum, gsumsq, gcnt, gmean, grstd, bp_r, h2, stream);

  // --- classifier ---
  pack_gemm(h2,     FEAT, c_w1, 128, c_b1, hidden,        128, NN, FEAT, 128, 1, bp_c1, stream);
  pack_gemm(hidden, 128,  c_w2, 49,  c_b2, (float*)d_out, 49,  NN, 128,  49,  0, bp_c2, stream);
}
